// SwinTransformer_5111011082415
// MI455X (gfx1250) — matrix operand order repaired and checked by reading, not compiled, not run on hardware
//
#include <hip/hip_runtime.h>
#include <hip/hip_bf16.h>
#include <math.h>

// ---------------------------------------------------------------------------
// CDNA5 (gfx1250) wave32 WMMA types & helpers
// ---------------------------------------------------------------------------
typedef __attribute__((ext_vector_type(16))) _Float16 v16h;
typedef __attribute__((ext_vector_type(8)))  _Float16 v8h;
typedef __attribute__((ext_vector_type(4)))  _Float16 v4h;
typedef __attribute__((ext_vector_type(8)))  float    v8f;
typedef __attribute__((ext_vector_type(4)))  float    v4f;

static __device__ __forceinline__ v8f wmma_f16(v16h a, v16h b, v8f c) {
    // D = A(16x32,f16) * B(32x16,f16) + C(16x16,f32)
    return __builtin_amdgcn_wmma_f32_16x16x32_f16(
        /*neg_a=*/false, a, /*neg_b=*/false, b,
        /*c_mod=*/(short)0, c, /*reuse_a=*/false, /*reuse_b=*/false);
}

static __device__ __forceinline__ v16h make_v16h(v8h lo, v8h hi) {
    v16h r;
#pragma unroll
    for (int i = 0; i < 8; ++i) { r[i] = lo[i]; r[i + 8] = hi[i]; }
    return r;
}

static __device__ __forceinline__ v4h cvt4(v4f x, float scale) {
    v4h h;
#pragma unroll
    for (int i = 0; i < 4; ++i) h[i] = (_Float16)(x[i] * scale);
    return h;
}

// A fragment: 16x32 f16 tile, stored row-major [m][k], row length `rowlen`.
// ISA layout: lane L holds M = mBase + (L&15); elems e<8 -> k = 8*(L>>4)+e,
// e>=8 -> k = 16 + 8*(L>>4) + (e-8).
static __device__ __forceinline__ v16h fragA(const _Float16* tile, int rowlen,
                                             int lane, int mBase, int kOff) {
    int m  = mBase + (lane & 15);
    int h8 = (lane >> 4) * 8;
    const _Float16* p = tile + (size_t)m * rowlen + kOff + h8;
    v8h lo = *(const v8h*)p;
    v8h hi = *(const v8h*)(p + 16);
    return make_v16h(lo, hi);
}

// B fragment: 32x16 f16 tile (KxN), stored "weight-row" style tile[n][k],
// row length `rowlen`. ISA layout: lane L holds N = nBase + (L&15);
// elems e<8 -> k = 8*(L>>4)+e, e>=8 -> k = 16 + 8*(L>>4) + (e-8).
static __device__ __forceinline__ v16h fragB(const _Float16* tile, int rowlen,
                                             int lane, int nBase, int kOff) {
    int n   = nBase + (lane & 31 & 15);
    int h16 = (lane >> 4) * 8;
    const _Float16* p = tile + (size_t)n * rowlen + kOff + h16;
    v8h lo = *(const v8h*)p;
    v8h hi = *(const v8h*)(p + 16);
    return make_v16h(lo, hi);
}

// ---------------------------------------------------------------------------
// Generic WMMA GEMM:  out[M,N] = act( A[M,K] * W[N,K]^T [+ bias] ) [+ residual]
// M % 128 == 0, N % 64 == 0, K % 32 == 0.  Block = 256 threads (8 waves),
// 128x64 output tile; wave grid 4x2, each wave owns a 32x32 quadrant
// (2x2 WMMA subtiles).  Staging: each thread owns 4 contiguous k ->
// global_load_b128 + ds_store_b64.  ACT: 0 = none, 1 = exact GELU.
// ---------------------------------------------------------------------------
template <int ACT, bool HAS_BIAS, bool HAS_RES>
__global__ __launch_bounds__(256) void gemm_wmma_kernel(
    const float* __restrict__ A, const float* __restrict__ W,
    const float* __restrict__ bias, const float* __restrict__ res,
    float* __restrict__ out, int M, int N, int K)
{
    __shared__ __align__(16) _Float16 As[128 * 32];
    __shared__ __align__(16) _Float16 Bs[64 * 32];

    const int tid  = threadIdx.x;
    const int lane = tid & 31;
    const int wv   = tid >> 5;          // 0..7
    const int tileN = blockIdx.x * 64;
    const int tileM = blockIdx.y * 128;
    const int mB = (wv & 3) * 32;       // 0,32,64,96
    const int nB = (wv >> 2) * 32;      // 0,32

    v8f acc00 = {}, acc01 = {}, acc10 = {}, acc11 = {};

    const int k4 = (tid & 7) * 4;       // 4 contiguous k per thread
    const int r0 = tid >> 3;            // 0..31

    for (int kb = 0; kb < K; kb += 32) {
        __syncthreads();
#pragma unroll
        for (int i = 0; i < 4; ++i) {   // A: 128 rows in 4 passes of 32
            int r = i * 32 + r0;
            v4f av = *(const v4f*)&A[(size_t)(tileM + r) * K + kb + k4];
            *(v4h*)&As[r * 32 + k4] = cvt4(av, 1.0f);
        }
#pragma unroll
        for (int i = 0; i < 2; ++i) {   // B: 64 rows in 2 passes of 32
            int r = i * 32 + r0;
            v4f wv4 = *(const v4f*)&W[(size_t)(tileN + r) * K + kb + k4];
            *(v4h*)&Bs[r * 32 + k4] = cvt4(wv4, 1.0f);
        }
        __syncthreads();
        v16h a0 = fragA(As, 32, lane, mB +  0, 0);
        v16h a1 = fragA(As, 32, lane, mB + 16, 0);
        v16h b0 = fragB(Bs, 32, lane, nB +  0, 0);
        v16h b1 = fragB(Bs, 32, lane, nB + 16, 0);
        acc00 = wmma_f16(a0, b0, acc00);
        acc01 = wmma_f16(a0, b1, acc01);
        acc10 = wmma_f16(a1, b0, acc10);
        acc11 = wmma_f16(a1, b1, acc11);
    }

    // D layout: lane L, vgpr v -> m = v + 8*(L>>4), n = L&15.
    const int mHalf = (lane >> 4) * 8;
    const int nLoc  = lane & 15;
    v8f accs[4] = {acc00, acc01, acc10, acc11};
#pragma unroll
    for (int s = 0; s < 4; ++s) {
        int mSub = (s >> 1) * 16, nSub = (s & 1) * 16;
        v8f acc = accs[s];
#pragma unroll
        for (int v = 0; v < 8; ++v) {
            int row = tileM + mB + mSub + mHalf + v;
            int col = tileN + nB + nSub + nLoc;
            float val = acc[v];
            if (HAS_BIAS) val += bias[col];
            if (ACT == 1) val = 0.5f * val * (1.0f + erff(val * 0.70710678118654752f));
            size_t o = (size_t)row * N + col;
            if (HAS_RES) val += res[o];
            out[o] = val;
        }
    }
}

// ---------------------------------------------------------------------------
// Fused window attention, one block per (window, batch, head). head_dim == 32.
// qkv: [B*H, 3C] (q | k | v).  out: [B*H, C].
// Handles token roll by `shift`, last-window shift mask, rel-pos bias.
// ---------------------------------------------------------------------------
__global__ __launch_bounds__(128) void attn_kernel(
    const float* __restrict__ qkv, const float* __restrict__ rpb,
    float* __restrict__ out, int Htok, int C, int nh, int shift)
{
    __shared__ __align__(16) _Float16 Qh[64 * 32];  // [token][dim]
    __shared__ __align__(16) _Float16 Kh[64 * 32];  // [token][dim]
    __shared__ __align__(16) _Float16 Vt[32 * 64];  // [dim][token] (transposed)
    __shared__ __align__(16) float    S [64 * 65];  // scores (padded)
    __shared__ __align__(16) _Float16 Ph[64 * 72];  // softmax probs (padded)

    const int tid  = threadIdx.x;
    const int lane = tid & 31;
    const int wv   = tid >> 5;
    const int win  = blockIdx.x;
    const int b    = blockIdx.y;
    const int head = blockIdx.z;
    const int nW   = Htok >> 6;
    const size_t C3 = (size_t)3 * C;

    // ---- load Q (scaled), K, V tiles: vectorized 4-wide ----
    const int c4 = (tid & 7) * 4;   // 4 contiguous head-dims per thread
    const int r0 = tid >> 3;        // 16 rows per pass
#pragma unroll
    for (int i = 0; i < 4; ++i) {
        int r = i * 16 + r0;                              // token in window
        int t = (win * 64 + r + shift) % Htok;            // rolled -> original
        size_t rowoff = ((size_t)b * Htok + t) * C3;
        int hc = head * 32 + c4;
        v4f qv = *(const v4f*)&qkv[rowoff + hc];
        v4f kv = *(const v4f*)&qkv[rowoff + (size_t)C + hc];
        v4f vv = *(const v4f*)&qkv[rowoff + (size_t)2 * C + hc];
        *(v4h*)&Qh[r * 32 + c4] = cvt4(qv, 0.17677669529663687f);  // 32^-0.5
        *(v4h*)&Kh[r * 32 + c4] = cvt4(kv, 1.0f);
#pragma unroll
        for (int j = 0; j < 4; ++j)
            Vt[(c4 + j) * 64 + r] = (_Float16)vv[j];
    }
    __syncthreads();

    // ---- S = Q * K^T : wave wv computes 16-row strip over all 64 columns ----
    {
        v16h a = fragA(Qh, 32, lane, 16 * wv, 0);
        int mHalf = (lane >> 4) * 8, nLoc = lane & 15;
#pragma unroll
        for (int s = 0; s < 4; ++s) {
            v16h bf = fragB(Kh, 32, lane, 16 * s, 0);
            v8f z = {};
            v8f acc = wmma_f16(a, bf, z);
#pragma unroll
            for (int v = 0; v < 8; ++v) {
                int i = 16 * wv + mHalf + v;
                int j = 16 * s + nLoc;
                S[i * 65 + j] = acc[v];
            }
        }
    }
    __syncthreads();

    // ---- row softmax with rel-pos bias + shift mask (threads 0..63) ----
    if (tid < 64) {
        int i = tid;
        bool maskw = (shift > 0) && (win == nW - 1);
        float mx = -1e30f;
        for (int j = 0; j < 64; ++j) {
            float v = S[i * 65 + j] + rpb[(i - j + 63) * nh + head];
            if (maskw && ((i < 32) != (j < 32))) v += -100.0f;
            S[i * 65 + j] = v;
            mx = fmaxf(mx, v);
        }
        float sum = 0.0f;
        for (int j = 0; j < 64; ++j) {
            float e = expf(S[i * 65 + j] - mx);
            S[i * 65 + j] = e;
            sum += e;
        }
        float inv = 1.0f / sum;
        for (int j = 0; j < 64; ++j)
            Ph[i * 72 + j] = (_Float16)(S[i * 65 + j] * inv);
    }
    __syncthreads();

    // ---- O = P * V : wave wv -> rows 16*wv..+15, cols 0..31, K = 64 ----
    v8f o0 = {}, o1 = {};
#pragma unroll
    for (int kk = 0; kk < 64; kk += 32) {
        v16h a  = fragA(Ph, 72, lane, 16 * wv, kk);
        v16h b0 = fragB(Vt, 64, lane,  0, kk);
        v16h b1 = fragB(Vt, 64, lane, 16, kk);
        o0 = wmma_f16(a, b0, o0);
        o1 = wmma_f16(a, b1, o1);
    }
    const int mHalf = (lane >> 4) * 8;
    const int nLoc  = lane & 15;
#pragma unroll
    for (int v = 0; v < 8; ++v) {
        int i = 16 * wv + mHalf + v;
        int t = (win * 64 + i + shift) % Htok;
        size_t row = ((size_t)b * Htok + t) * C + head * 32;
        out[row + nLoc]      = o0[v];
        out[row + 16 + nLoc] = o1[v];
    }
}

// ---------------------------------------------------------------------------
// LayerNorm over last dim C, one 128-thread block per row.
// ---------------------------------------------------------------------------
__global__ __launch_bounds__(128) void ln_kernel(
    const float* __restrict__ src, float* __restrict__ dst,
    const float* __restrict__ g, const float* __restrict__ bta, int C)
{
    __shared__ float red[128];
    const int row = blockIdx.x, tid = threadIdx.x;
    const float* x = src + (size_t)row * C;
    float s = 0.0f;
    for (int c = tid; c < C; c += 128) s += x[c];
    red[tid] = s; __syncthreads();
    for (int st = 64; st > 0; st >>= 1) { if (tid < st) red[tid] += red[tid + st]; __syncthreads(); }
    float mean = red[0] / C;
    __syncthreads();
    float vs = 0.0f;
    for (int c = tid; c < C; c += 128) { float d = x[c] - mean; vs += d * d; }
    red[tid] = vs; __syncthreads();
    for (int st = 64; st > 0; st >>= 1) { if (tid < st) red[tid] += red[tid + st]; __syncthreads(); }
    float inv = rsqrtf(red[0] / C + 1e-5f);
    float* y = dst + (size_t)row * C;
    for (int c = tid; c < C; c += 128) y[c] = (x[c] - mean) * inv * g[c] + bta[c];
}

// ---------------------------------------------------------------------------
// Patch embedding: Conv1d(k = stride = 4): out[b,n,e] = sum_{c,p} x*w + b
// ---------------------------------------------------------------------------
__global__ __launch_bounds__(256) void patch_embed_kernel(
    const float* __restrict__ x, const float* __restrict__ w,
    const float* __restrict__ bias, float* __restrict__ out,
    int L, int nP, int E)
{
    size_t idx = (size_t)blockIdx.x * 256 + threadIdx.x;   // over B*nP*E
    int e = (int)(idx % E);
    size_t bn = idx / E;
    int n = (int)(bn % nP);
    int b = (int)(bn / nP);
    const float* xb = x + (size_t)b * 3 * L + (size_t)n * 4;
    const float* we = w + (size_t)e * 12;
    float s = bias[e];
#pragma unroll
    for (int cc = 0; cc < 3; ++cc) {
        v4f xv = *(const v4f*)&xb[(size_t)cc * L];
#pragma unroll
        for (int p = 0; p < 4; ++p)
            s += xv[p] * we[cc * 4 + p];
    }
    out[idx] = s;
}

// ---------------------------------------------------------------------------
// Patch merge gather: [B,H,C] -> [B,H/2,2C]  (concat even | odd tokens)
// ---------------------------------------------------------------------------
__global__ __launch_bounds__(256) void merge_gather_kernel(
    const float* __restrict__ h, float* __restrict__ y, int Htok, int C)
{
    size_t idx = (size_t)blockIdx.x * 256 + threadIdx.x;  // over B*(H/2)*(2C)
    int C2 = 2 * C;
    int c = (int)(idx % C2);
    size_t ng = idx / C2;
    int H2 = Htok / 2;
    int n = (int)(ng % H2);
    int b = (int)(ng / H2);
    size_t src = (c < C) ? (((size_t)b * Htok + 2 * n) * C + c)
                         : (((size_t)b * Htok + 2 * n + 1) * C + (c - C));
    y[idx] = h[src];
}

// ---------------------------------------------------------------------------
// Mean-pool over tokens:  pooled[b,c] = mean_t hn[b,t,c]
// ---------------------------------------------------------------------------
__global__ __launch_bounds__(256) void pool_kernel(
    const float* __restrict__ hn, float* __restrict__ pooled, int Htok, int C)
{
    int idx = blockIdx.x * 256 + threadIdx.x;  // over B*C
    int b = idx / C, c = idx % C;
    float s = 0.0f;
    for (int t = 0; t < Htok; ++t) s += hn[((size_t)b * Htok + t) * C + c];
    pooled[idx] = s / (float)Htok;
}

// ---------------------------------------------------------------------------
// Classifier head: out[b,o] = pooled[b,:] . head_w[o,:] + head_b[o]
// ---------------------------------------------------------------------------
__global__ __launch_bounds__(256) void head_kernel(
    const float* __restrict__ pooled, const float* __restrict__ w,
    const float* __restrict__ bias, float* __restrict__ out, int C, int NC, int total)
{
    int idx = blockIdx.x * 256 + threadIdx.x;
    if (idx >= total) return;
    int b = idx / NC, o = idx % NC;
    float s = bias[o];
    const float* pw = w + (size_t)o * C;
    const float* pp = pooled + (size_t)b * C;
    for (int c = 0; c < C; ++c) s += pp[c] * pw[c];
    out[idx] = s;
}

// ---------------------------------------------------------------------------
// Host orchestration
// ---------------------------------------------------------------------------
extern "C" void kernel_launch(void* const* d_in, const int* in_sizes, int n_in,
                              void* d_out, int out_size, void* d_ws, size_t ws_size,
                              hipStream_t stream)
{
    (void)in_sizes; (void)out_size; (void)ws_size;
    if (n_in < 174 || !d_ws) return;

    const int BATCH = 8, SEQ = 16384, E0 = 128, WIN = 64, NCLS = 1000;
    const int depths[4] = {2, 2, 6, 2};
    const int nheads[4] = {4, 8, 16, 32};

    const float* x     = (const float*)d_in[0];
    const float* pe_w  = (const float*)d_in[1];
    const float* pe_b  = (const float*)d_in[2];
    const float* pe_ng = (const float*)d_in[3];
    const float* pe_nb = (const float*)d_in[4];
    // blocks: indices 5 .. 160 (12 blocks x 13 tensors, insertion order)
    // merges: indices 161..169 (3 x {n_g, n_b, red_w})
    const float* fn_g   = (const float*)d_in[170];
    const float* fn_b   = (const float*)d_in[171];
    const float* head_w = (const float*)d_in[172];
    const float* head_b = (const float*)d_in[173];

    float* ws = (float*)d_ws;
    const size_t NF = (size_t)4 * 1024 * 1024;   // 4M floats per unit buffer
    float* bufH   = ws;            // current activations    [M, C]
    float* bufN   = ws + NF;       // LN output              [M, C]
    float* bufA   = ws + 2 * NF;   // attention out / pooled [M, C]
    float* bufBig = ws + 3 * NF;   // qkv [M,3C] / MLP hidden [M,4C] / merged

    // mode: 0 = bias, 1 = bias+gelu, 2 = bias+residual, 3 = plain
    auto gemm = [&](const float* A, const float* W, const float* bias,
                    const float* res, float* out, int M, int N, int K, int mode) {
        dim3 g(N / 64, M / 128);
        if (mode == 0)
            gemm_wmma_kernel<0, true, false><<<g, 256, 0, stream>>>(A, W, bias, res, out, M, N, K);
        else if (mode == 1)
            gemm_wmma_kernel<1, true, false><<<g, 256, 0, stream>>>(A, W, bias, res, out, M, N, K);
        else if (mode == 2)
            gemm_wmma_kernel<0, true, true ><<<g, 256, 0, stream>>>(A, W, bias, res, out, M, N, K);
        else
            gemm_wmma_kernel<0, false, false><<<g, 256, 0, stream>>>(A, W, bias, res, out, M, N, K);
    };

    // ---- patch embed + LN ----
    const int nP = SEQ / 4;            // 4096 tokens per image
    const int M0 = BATCH * nP;         // 32768
    patch_embed_kernel<<<(M0 * E0) / 256, 256, 0, stream>>>(x, pe_w, pe_b, bufN, SEQ, nP, E0);
    ln_kernel<<<M0, 128, 0, stream>>>(bufN, bufH, pe_ng, pe_nb, E0);

    int Htok = nP;
    int C = E0;
    int bp = 5;
    for (int st = 0; st < 4; ++st) {
        const int M  = BATCH * Htok;
        const int nh = nheads[st];
        for (int blk = 0; blk < depths[st]; ++blk) {
            const float* n1_g   = (const float*)d_in[bp + 0];
            const float* n1_b   = (const float*)d_in[bp + 1];
            const float* qkv_w  = (const float*)d_in[bp + 2];
            const float* qkv_b  = (const float*)d_in[bp + 3];
            const float* proj_w = (const float*)d_in[bp + 4];
            const float* proj_b = (const float*)d_in[bp + 5];
            const float* rpb    = (const float*)d_in[bp + 6];
            const float* n2_g   = (const float*)d_in[bp + 7];
            const float* n2_b   = (const float*)d_in[bp + 8];
            const float* fc1_w  = (const float*)d_in[bp + 9];
            const float* fc1_b  = (const float*)d_in[bp + 10];
            const float* fc2_w  = (const float*)d_in[bp + 11];
            const float* fc2_b  = (const float*)d_in[bp + 12];
            bp += 13;
            const int shift = (blk & 1) ? (WIN / 2) : 0;

            // x + attn(LN(x))
            ln_kernel<<<M, 128, 0, stream>>>(bufH, bufN, n1_g, n1_b, C);
            gemm(bufN, qkv_w, qkv_b, nullptr, bufBig, M, 3 * C, C, 0);
            dim3 ag(Htok / 64, BATCH, nh);
            attn_kernel<<<ag, 128, 0, stream>>>(bufBig, rpb, bufA, Htok, C, nh, shift);
            gemm(bufA, proj_w, proj_b, bufH, bufH, M, C, C, 2);

            // x + fc2(gelu(fc1(LN(x))))
            ln_kernel<<<M, 128, 0, stream>>>(bufH, bufN, n2_g, n2_b, C);
            gemm(bufN, fc1_w, fc1_b, nullptr, bufBig, M, 4 * C, C, 1);
            gemm(bufBig, fc2_w, fc2_b, bufH, bufH, M, C, 4 * C, 2);
        }
        if (st < 3) {
            const float* mg = (const float*)d_in[161 + st * 3 + 0];
            const float* mb = (const float*)d_in[161 + st * 3 + 1];
            const float* mw = (const float*)d_in[161 + st * 3 + 2];
            const int M2 = M / 2, C2 = 2 * C;
            size_t tot = (size_t)M2 * C2;
            merge_gather_kernel<<<(unsigned)(tot / 256), 256, 0, stream>>>(bufH, bufBig, Htok, C);
            ln_kernel<<<M2, 128, 0, stream>>>(bufBig, bufN, mg, mb, C2);
            gemm(bufN, mw, nullptr, nullptr, bufH, M2, C2, C2, 3);
            Htok /= 2;
            C *= 2;
        }
    }

    // ---- final LN, pool, head ----
    const int Mf = BATCH * Htok;       // 4096
    ln_kernel<<<Mf, 128, 0, stream>>>(bufH, bufN, fn_g, fn_b, C);
    pool_kernel<<<(BATCH * C) / 256, 256, 0, stream>>>(bufN, bufA, Htok, C);
    int total = BATCH * NCLS;
    head_kernel<<<(total + 255) / 256, 256, 0, stream>>>(bufA, head_w, head_b,
                                                         (float*)d_out, C, NCLS, total);
}
